// GCN_69630009802900
// MI455X (gfx1250) — matrix operand order repaired and checked by reading, not compiled, not run on hardware
//
#include <hip/hip_runtime.h>
#include <hip/hip_bf16.h>
#include <math.h>

// ---------------------------------------------------------------------------
// GCN: 3x (GEMM -> edge gather/scale/scatter-add -> bias/act), log_softmax.
// Layer-1 GEMM (100000x512 @ 512x32) uses v_wmma_f32_16x16x32_f16.
// Each wave owns a 16x32 output tile: one A fragment feeds two WMMAs,
// halving HBM reads of x (the dominant stream).
// ---------------------------------------------------------------------------

#define NN 100000
#define IN_DIM 512
#define HD1 32
#define HD2 16
#define OD  40

typedef __attribute__((ext_vector_type(16))) _Float16 v16h;
typedef __attribute__((ext_vector_type(8)))  _Float16 v8h;
typedef __attribute__((ext_vector_type(8)))  float    v8f;
typedef __attribute__((ext_vector_type(4)))  float    v4f;

// ---------------- degree / normalization ----------------
__global__ void k_init_deg(float* deg, int n) {
    int i = blockIdx.x * blockDim.x + threadIdx.x;
    if (i < n) deg[i] = 1.0f;                      // self-loop contribution
}
__global__ void k_count_deg(const int* __restrict__ dst, float* deg, int e) {
    int i = blockIdx.x * blockDim.x + threadIdx.x;
    if (i < e) atomicAdd(&deg[__builtin_nontemporal_load(dst + i)], 1.0f);
}
__global__ void k_rsqrt(float* deg, int n) {
    int i = blockIdx.x * blockDim.x + threadIdx.x;
    if (i < n) deg[i] = rsqrtf(deg[i]);            // deg >= 1 always
}

// ---------------- layer-1 GEMM via WMMA ----------------
// 128 threads = 4 waves; each wave: 16 rows x 32 cols (2 WMMAs / K-step).
// Row guard is wave-uniform so EXEC stays all-1s around WMMA.
#define WPAD (IN_DIM + 8)   // +16B: keeps 16B align, spreads LDS banks
__global__ void __launch_bounds__(128)
k_gemm1_wmma(const float* __restrict__ x, const float* __restrict__ W,
             float* __restrict__ h, int n)
{
    __shared__ _Float16 WlT[HD1][WPAD];            // W1 transposed, f16, ~33 KB
    const int tid = threadIdx.x;
    for (int i = tid; i < IN_DIM * HD1; i += 128) {
        int k = i / HD1, c = i % HD1;              // W row-major [512][32]
        WlT[c][k] = (_Float16)W[i];
    }
    __syncthreads();

    const int wave = tid >> 5;
    const int lane = tid & 31;
    const int row0 = (blockIdx.x * 4 + wave) * 16; // wave-uniform
    if (row0 >= n) return;                         // uniform exit: EXEC intact

    const int m = lane & 15;                       // row (A) / col (B/D)
    const int g = lane >> 4;                       // lane group

    const float* __restrict__ xrow = x + (size_t)(row0 + m) * IN_DIM;
    v8f acc0 = {};                                 // cols 0..15
    v8f acc1 = {};                                 // cols 16..31

    for (int kt = 0; kt < IN_DIM; kt += 32) {
        // ---- A fragment: 16-bit A 16x32 layout (ISA 7.12.2) ----
        // lane group g: K = kb..kb+7 -> a[0..7], K = kb+16..kb+23 -> a[8..15]
        const int kb = kt + g * 8;
        const v4f p0 = __builtin_nontemporal_load((const v4f*)(xrow + kb));
        const v4f p1 = __builtin_nontemporal_load((const v4f*)(xrow + kb + 4));
        const v4f q0 = __builtin_nontemporal_load((const v4f*)(xrow + kb + 16));
        const v4f q1 = __builtin_nontemporal_load((const v4f*)(xrow + kb + 20));
        v16h a;
        #pragma unroll
        for (int j = 0; j < 4; ++j) {
            a[j]      = (_Float16)p0[j];
            a[4 + j]  = (_Float16)p1[j];
            a[8 + j]  = (_Float16)q0[j];
            a[12 + j] = (_Float16)q1[j];
        }

        // ---- B fragments: 32x16, row striped across lanes ----
        // lane group g: K = kt+g*8..+7 -> b[0..7], K = kt+16+g*8..+7 -> b[8..15]
        const _Float16* wp0 = &WlT[m][kt + g * 8];
        const _Float16* wp1 = &WlT[16 + m][kt + g * 8];
        const v8h b0lo = *(const v8h*)(wp0);
        const v8h b0hi = *(const v8h*)(wp0 + 16);
        const v8h b1lo = *(const v8h*)(wp1);
        const v8h b1hi = *(const v8h*)(wp1 + 16);
        v16h b0, b1;
        #pragma unroll
        for (int j = 0; j < 8; ++j) {
            b0[j] = b0lo[j]; b0[j + 8] = b0hi[j];
            b1[j] = b1lo[j]; b1[j + 8] = b1hi[j];
        }

        acc0 = __builtin_amdgcn_wmma_f32_16x16x32_f16(
            false, a, false, b0, (short)0, acc0, false, false);
        acc1 = __builtin_amdgcn_wmma_f32_16x16x32_f16(
            false, a, false, b1, (short)0, acc1, false, false);
    }

    // ---- D: 32-bit C/D 16x16 layout: acc[v] -> row (v + 8*g), col m ----
    #pragma unroll
    for (int v = 0; v < 8; ++v) {
        float* orow = h + (size_t)(row0 + v + 8 * g) * HD1;
        orow[m]      = acc0[v];
        orow[16 + m] = acc1[v];
    }
}

// ---------------- aggregation ----------------
// agg[i][:] = dis[i]^2 * h[i][:]   (self-loop term; also initializes buffer)
__global__ void k_init_agg(const float* __restrict__ h, const float* __restrict__ dis,
                           float* __restrict__ agg, int n, int F)
{
    int i = blockIdx.x * blockDim.x + threadIdx.x;
    if (i >= n * F) return;
    int node = i / F;
    float d = dis[node];
    agg[i] = d * d * h[i];
}

template <int F>
__global__ void k_scatter(const int* __restrict__ src, const int* __restrict__ dst,
                          const float* __restrict__ dis, const float* __restrict__ h,
                          float* __restrict__ agg, int e)
{
    int i = blockIdx.x * blockDim.x + threadIdx.x;
    if (i >= e) return;
    int s = __builtin_nontemporal_load(src + i);   // edge lists: stream-once
    int d = __builtin_nontemporal_load(dst + i);
    float nm = dis[s] * dis[d];                    // dis: L2-resident, reuse
    const v4f* __restrict__ hp = (const v4f*)(h + (size_t)s * F);
    float* __restrict__ ap = agg + (size_t)d * F;
    #pragma unroll
    for (int c = 0; c < F / 4; ++c) {
        v4f v = hp[c];
        atomicAdd(ap + 4 * c + 0, v[0] * nm);
        atomicAdd(ap + 4 * c + 1, v[1] * nm);
        atomicAdd(ap + 4 * c + 2, v[2] * nm);
        atomicAdd(ap + 4 * c + 3, v[3] * nm);
    }
}

__global__ void k_bias_relu(float* __restrict__ a, const float* __restrict__ b,
                            int total, int F)
{
    int i = blockIdx.x * blockDim.x + threadIdx.x;
    if (i >= total) return;
    a[i] = fmaxf(a[i] + b[i % F], 0.0f);
}

// ---------------- small dense layers (scalar) ----------------
template <int K, int N>
__global__ void __launch_bounds__(256)
k_gemm_small(const float* __restrict__ in, const float* __restrict__ W,
             float* __restrict__ out, int n)
{
    __shared__ float Ws[K * N];
    for (int i = threadIdx.x; i < K * N; i += 256) Ws[i] = W[i];
    __syncthreads();
    int node = blockIdx.x * blockDim.x + threadIdx.x;
    if (node >= n) return;
    float xi[K];
    #pragma unroll
    for (int k = 0; k < K; ++k) xi[k] = in[(size_t)node * K + k];
    #pragma unroll
    for (int j = 0; j < N; ++j) {
        float s = 0.0f;
        #pragma unroll
        for (int k = 0; k < K; ++k) s = fmaf(xi[k], Ws[k * N + j], s);
        out[(size_t)node * N + j] = s;
    }
}

// ---------------- bias + log_softmax over 40 classes ----------------
__global__ void k_logsoftmax40(float* __restrict__ io, const float* __restrict__ b3,
                               int n)
{
    int node = blockIdx.x * blockDim.x + threadIdx.x;
    if (node >= n) return;
    float* row = io + (size_t)node * OD;
    float r[OD];
    float mx = -3.402823466e+38f;
    #pragma unroll
    for (int j = 0; j < OD; ++j) { r[j] = row[j] + b3[j]; mx = fmaxf(mx, r[j]); }
    float s = 0.0f;
    #pragma unroll
    for (int j = 0; j < OD; ++j) s += expf(r[j] - mx);
    float l = logf(s);
    #pragma unroll
    for (int j = 0; j < OD; ++j) row[j] = r[j] - mx - l;
}

// ---------------------------------------------------------------------------
extern "C" void kernel_launch(void* const* d_in, const int* in_sizes, int n_in,
                              void* d_out, int out_size, void* d_ws, size_t ws_size,
                              hipStream_t stream)
{
    const float* x  = (const float*)d_in[0];
    const int*   ei = (const int*)  d_in[1];
    const float* W1 = (const float*)d_in[2];
    const float* b1 = (const float*)d_in[3];
    const float* W2 = (const float*)d_in[4];
    const float* b2 = (const float*)d_in[5];
    const float* W3 = (const float*)d_in[6];
    const float* b3 = (const float*)d_in[7];
    float* out = (float*)d_out;

    const int n = in_sizes[0] / IN_DIM;        // 100000 (multiple of 16)
    const int E = in_sizes[1] / 2;             // 6.4M
    const int* src = ei;
    const int* dst = ei + E;

    // workspace: dis[n] | bufA[n*40] | bufB[n*40]  (~32.4 MB)
    float* dis  = (float*)d_ws;
    float* bufA = dis + ((n + 255) & ~255);
    float* bufB = bufA + (size_t)n * OD;

    const int B  = 256;
    const int gN = (n + B - 1) / B;
    const int gE = (E + B - 1) / B;

    // degree -> dis = deg^{-1/2}
    k_init_deg <<<gN, B, 0, stream>>>(dis, n);
    k_count_deg<<<gE, B, 0, stream>>>(dst, dis, E);
    k_rsqrt    <<<gN, B, 0, stream>>>(dis, n);

    // ---- layer 1: h1 = x @ W1 (WMMA), aggregate, +b1, relu ----
    const int nWaveTiles = n / 16;                 // 6250
    const int gG = (nWaveTiles + 3) / 4;           // 4 waves per block
    k_gemm1_wmma<<<gG, 128, 0, stream>>>(x, W1, bufA, n);
    k_init_agg  <<<(n * HD1 + B - 1) / B, B, 0, stream>>>(bufA, dis, bufB, n, HD1);
    k_scatter<HD1><<<gE, B, 0, stream>>>(src, dst, dis, bufA, bufB, E);
    k_bias_relu <<<(n * HD1 + B - 1) / B, B, 0, stream>>>(bufB, b1, n * HD1, HD1);

    // ---- layer 2: h2 = relu1 @ W2, aggregate, +b2, relu ----
    k_gemm_small<HD1, HD2><<<gN, B, 0, stream>>>(bufB, W2, bufA, n);
    k_init_agg  <<<(n * HD2 + B - 1) / B, B, 0, stream>>>(bufA, dis, bufB, n, HD2);
    k_scatter<HD2><<<gE, B, 0, stream>>>(src, dst, dis, bufA, bufB, E);
    k_bias_relu <<<(n * HD2 + B - 1) / B, B, 0, stream>>>(bufB, b2, n * HD2, HD2);

    // ---- layer 3: h3 = relu2 @ W3, aggregate into d_out, +b3, log_softmax ----
    k_gemm_small<HD2, OD><<<gN, B, 0, stream>>>(bufB, W3, bufA, n);
    k_init_agg  <<<(n * OD + B - 1) / B, B, 0, stream>>>(bufA, dis, out, n, OD);
    k_scatter<OD><<<gE, B, 0, stream>>>(src, dst, dis, bufA, out, E);
    k_logsoftmax40<<<gN, B, 0, stream>>>(out, b3, n);
}
